// HeteroGNNWithSoftMinTopK_89644557402635
// MI455X (gfx1250) — compile-verified, matrix operand order checked by reading, not run on hardware
//
#include <hip/hip_runtime.h>
#include <math.h>

typedef __attribute__((ext_vector_type(16))) _Float16 v16h;
typedef __attribute__((ext_vector_type(8)))  _Float16 v8h;
typedef __attribute__((ext_vector_type(8)))  float    v8f;
typedef __attribute__((ext_vector_type(4)))  float    v4f;

constexpr int   HD   = 128;   // hidden dim H
constexpr float TAUF = 0.1f;

// ---------- order-preserving float<->uint transform (for atomic max / radix select) ----------
__device__ __forceinline__ unsigned f2ord(float f) {
  unsigned u = __float_as_uint(f);
  return (u & 0x80000000u) ? ~u : (u | 0x80000000u);
}
__device__ __forceinline__ float ord2f(unsigned o) {
  unsigned u = (o & 0x80000000u) ? (o & 0x7fffffffu) : ~o;
  return __uint_as_float(u);
}

// ---------- fp32 -> fp16 cast ----------
__global__ void cast_f16_k(const float* __restrict__ in, _Float16* __restrict__ out, long n) {
  long i = (long)blockIdx.x * blockDim.x + threadIdx.x;
  if (i < n) out[i] = (_Float16)in[i];
}

// ---------- zero agg/denom/segmax ----------
__global__ void zero_seg_k(float* __restrict__ agg, float* __restrict__ denom,
                           unsigned* __restrict__ segmax, int n_dst) {
  long i = (long)blockIdx.x * blockDim.x + threadIdx.x;
  long tot = (long)n_dst * HD;
  if (i < tot) agg[i] = 0.0f;
  if (i < n_dst) { denom[i] = 0.0f; segmax[i] = 0u; }
}

// =====================================================================================
// Pass 1: per-edge 2-layer MLP via WMMA. One wave32 handles a 16-edge tile.
//   x = [srcF[s], dstF[d]]  (16 x 256 f16)   y = relu(x@W1+b1)   m = y@W2+b2 (f32)
// Stores m to m_buf, neg = -||m||/tau to neg_buf, atomicMax ordered neg into segmax[dst].
// =====================================================================================
__launch_bounds__(256)
__global__ void edge_mlp_pass1(
    const _Float16* __restrict__ srcF, const _Float16* __restrict__ dstF,
    const int* __restrict__ sidx, const int* __restrict__ didx,
    const float* __restrict__ W1, const float* __restrict__ b1,
    const float* __restrict__ W2, const float* __restrict__ b2,
    float* __restrict__ m_buf, float* __restrict__ neg_buf,
    unsigned* __restrict__ segmax, int E)
{
  // Swizzled B-operand storage: tile (kt,nt) -> 32 lanes x 16 halves contiguous per lane.
  __shared__ __align__(32) _Float16 w1s[64 * 512];      // 64 KB  (8 ksteps x 8 ntiles)
  __shared__ __align__(32) _Float16 w2s[32 * 512];      // 32 KB  (4 ksteps x 8 ntiles)
  __shared__ __align__(32) _Float16 ytile[8][16 * HD];  // 32 KB  relu output per wave

  const int tid = threadIdx.x;

  // Cooperative fill of W1/W2 into the per-lane WMMA B layout (f32 -> f16 on the fly).
  for (int i = tid; i < 64 * 512; i += 256) {
    int tile = i >> 9, pos = i & 511;
    int ln = pos >> 4, s = pos & 15;
    int kt = tile >> 3, nt = tile & 7;
    int hh = ln >> 4, n = ln & 15;
    int k = kt * 32 + ((s < 8) ? (8 * hh + s) : (16 + 8 * hh + (s - 8)));
    w1s[i] = (_Float16)W1[k * HD + nt * 16 + n];
  }
  for (int i = tid; i < 32 * 512; i += 256) {
    int tile = i >> 9, pos = i & 511;
    int ln = pos >> 4, s = pos & 15;
    int kt = tile >> 3, nt = tile & 7;
    int hh = ln >> 4, n = ln & 15;
    int k = kt * 32 + ((s < 8) ? (8 * hh + s) : (16 + 8 * hh + (s - 8)));
    w2s[i] = (_Float16)W2[k * HD + nt * 16 + n];
  }
  __syncthreads();

  const int wave = tid >> 5;
  const int lane = tid & 31;
  const int hh = lane >> 4;    // which half of the wave
  const int mr = lane & 15;    // row (edge) index within the 16-edge tile

  const long base = ((long)blockIdx.x * 8 + wave) * 16;
  long e_m = base + mr; if (e_m >= E) e_m = (long)E - 1;   // clamp: WMMA needs full EXEC
  const int si = sidx[e_m];
  const int di = didx[e_m];
  const v8h* __restrict__ srow = (const v8h*)(srcF + (size_t)si * HD);
  const v8h* __restrict__ drow = (const v8h*)(dstF + (size_t)di * HD);

  // ---------------- layer 1: [16x256] @ [256x128] ----------------
  v8f acc[8] = {};
  #pragma unroll
  for (int kt = 0; kt < 8; ++kt) {
    const v8h* row = (kt < 4) ? srow : drow;
    const int kb = (kt & 3) * 4;
    v8h alo = row[kb + hh];
    v8h ahi = row[kb + 2 + hh];
    v16h a = __builtin_shufflevector(alo, ahi, 0,1,2,3,4,5,6,7,8,9,10,11,12,13,14,15);
    #pragma unroll
    for (int nt = 0; nt < 8; ++nt) {
      v16h b = *(const v16h*)&w1s[(kt * 8 + nt) * 512 + lane * 16];
      acc[nt] = __builtin_amdgcn_wmma_f32_16x16x32_f16(
          false, a, false, b, (short)0, acc[nt], false, false);
    }
  }

  // bias + relu, stage through LDS to re-layout C (row=j+8*hh, col=nt*16+mr) into A format
  _Float16* yt = &ytile[wave][0];
  #pragma unroll
  for (int nt = 0; nt < 8; ++nt) {
    float bias = b1[nt * 16 + mr];
    #pragma unroll
    for (int j = 0; j < 8; ++j) {
      float v = acc[nt][j] + bias;
      v = v > 0.0f ? v : 0.0f;
      yt[(j + 8 * hh) * HD + nt * 16 + mr] = (_Float16)v;
    }
  }
  __syncthreads();

  // ---------------- layer 2: [16x128] @ [128x128] ----------------
  v8f acc2[8] = {};
  const v8h* __restrict__ yrow = (const v8h*)&yt[mr * HD];
  #pragma unroll
  for (int kt = 0; kt < 4; ++kt) {
    v8h alo = yrow[kt * 4 + hh];
    v8h ahi = yrow[kt * 4 + 2 + hh];
    v16h a = __builtin_shufflevector(alo, ahi, 0,1,2,3,4,5,6,7,8,9,10,11,12,13,14,15);
    #pragma unroll
    for (int nt = 0; nt < 8; ++nt) {
      v16h b = *(const v16h*)&w2s[(kt * 8 + nt) * 512 + lane * 16];
      acc2[nt] = __builtin_amdgcn_wmma_f32_16x16x32_f16(
          false, a, false, b, (short)0, acc2[nt], false, false);
    }
  }

  // bias, store m, accumulate per-row sum of squares
  float rowsq[8];
  #pragma unroll
  for (int j = 0; j < 8; ++j) rowsq[j] = 0.0f;
  #pragma unroll
  for (int nt = 0; nt < 8; ++nt) {
    float bias2 = b2[nt * 16 + mr];
    #pragma unroll
    for (int j = 0; j < 8; ++j) {
      float v = acc2[nt][j] + bias2;
      long e = base + j + 8 * hh;
      if (e < E) m_buf[e * HD + nt * 16 + mr] = v;
      rowsq[j] += v * v;
    }
  }
  // reduce each row's sumsq across the 16 lanes of its half-wave
  #pragma unroll
  for (int j = 0; j < 8; ++j) {
    float s = rowsq[j];
    s += __shfl_xor(s, 1, 32);
    s += __shfl_xor(s, 2, 32);
    s += __shfl_xor(s, 4, 32);
    s += __shfl_xor(s, 8, 32);
    rowsq[j] = s;   // lane (hh,mr) now holds total for row (j + 8*hh)
  }
  // lane whose mr equals its own row index finalizes that edge (its si/di match)
  #pragma unroll
  for (int j = 0; j < 8; ++j) {
    if (mr == j + 8 * hh) {
      long e = base + mr;
      if (e < E) {
        float nv = -sqrtf(rowsq[j]) * (1.0f / TAUF);
        neg_buf[e] = nv;
        atomicMax(&segmax[di], f2ord(nv));
      }
    }
  }
}

// ---------- Pass 2: softmin scatter. 32 threads per edge, 4 cols each ----------
__global__ void edge_scatter(const float* __restrict__ m_buf, const float* __restrict__ neg_buf,
                             const unsigned* __restrict__ segmax, const int* __restrict__ didx,
                             float* __restrict__ agg, float* __restrict__ denom, int E)
{
  long t = (long)blockIdx.x * blockDim.x + threadIdx.x;
  long e = t >> 5;
  int ln = threadIdx.x & 31;
  if (e >= E) return;
  int d = didx[e];
  float w = __expf(neg_buf[e] - ord2f(segmax[d]));
  if (ln == 0) atomicAdd(&denom[d], w);
  v4f mv = *(const v4f*)&m_buf[e * HD + ln * 4];
  float* ap = &agg[(long)d * HD + ln * 4];
  atomicAdd(ap + 0, mv[0] * w);
  atomicAdd(ap + 1, mv[1] * w);
  atomicAdd(ap + 2, mv[2] * w);
  atomicAdd(ap + 3, mv[3] * w);
}

__global__ void normalize_k(float* __restrict__ agg, const float* __restrict__ denom, int n_dst) {
  long t = (long)blockIdx.x * blockDim.x + threadIdx.x;
  if (t >= (long)n_dst * HD) return;
  float d = denom[t >> 7];
  agg[t] = d > 0.0f ? agg[t] / d : 0.0f;
}

// ---------- TopK pool: scores + exact 4-level radix select on ordered uints ----------
__global__ void pool_score(const float* __restrict__ rr, const float* __restrict__ w,
                           const float* __restrict__ b, unsigned* __restrict__ sc, int n) {
  int i = blockIdx.x * blockDim.x + threadIdx.x;
  if (i >= n) return;
  float s = b[0];
  const v4f* row = (const v4f*)(rr + (long)i * HD);
  const v4f* wv = (const v4f*)w;
  #pragma unroll 4
  for (int h = 0; h < HD / 4; ++h) {
    v4f r = row[h], ww = wv[h];
    s += r[0] * ww[0] + r[1] * ww[1] + r[2] * ww[2] + r[3] * ww[3];
  }
  sc[i] = f2ord(s);
}

__global__ void sel_init(unsigned* __restrict__ counts, unsigned* __restrict__ sel, unsigned K) {
  int i = threadIdx.x;
  if (i < 256) counts[i] = 0u;
  if (i == 0) { sel[0] = 0u; sel[1] = K; sel[2] = 0u; }
}

__global__ void topk_hist(const unsigned* __restrict__ sc, unsigned* __restrict__ counts,
                          const unsigned* __restrict__ sel, int n, int shift, unsigned hi_mask) {
  int i = blockIdx.x * blockDim.x + threadIdx.x;
  if (i >= n) return;
  unsigned o = sc[i];
  if ((o & hi_mask) == (sel[0] & hi_mask))
    atomicAdd(&counts[(o >> shift) & 255u], 1u);
}

__global__ void topk_select(unsigned* __restrict__ counts, unsigned* __restrict__ sel, int shift) {
  unsigned k = sel[1];
  int bin = 255;
  for (; bin > 0; --bin) {
    unsigned c = counts[bin];
    if (k <= c) break;
    k -= c;
  }
  sel[0] |= ((unsigned)bin << shift);
  sel[1] = k;
  for (int i = 0; i < 256; ++i) counts[i] = 0u;
}

__global__ void topk_flags(const unsigned* __restrict__ sc, unsigned* __restrict__ sel,
                           unsigned* __restrict__ keep, int n) {
  int i = blockIdx.x * blockDim.x + threadIdx.x;
  if (i >= n) return;
  unsigned o = sc[i], thr = sel[0];
  unsigned kp = 0u;
  if (o > thr) kp = 1u;
  else if (o == thr) kp = (atomicAdd(&sel[2], 1u) < sel[1]) ? 1u : 0u;
  keep[i] = kp;
}

__global__ void topk_apply(const float* __restrict__ in, const unsigned* __restrict__ keep,
                           float* __restrict__ out, int n) {
  long t = (long)blockIdx.x * blockDim.x + threadIdx.x;
  if (t >= (long)n * HD) return;
  out[t] = keep[t >> 7] ? in[t] : 0.0f;
}

// ---------- readout ----------
__global__ void readout_init(float* __restrict__ out, const float* __restrict__ rr,
                             const int* __restrict__ sink) {
  int h = threadIdx.x;
  if (h < HD) out[h] = rr[(long)sink[0] * HD + h];
  else if (h < 2 * HD) out[h] = 0.0f;
}

__global__ void readout_sum(const float* __restrict__ tile, float* __restrict__ out, int n) {
  int h = threadIdx.x;  // 128 threads
  float s = 0.0f;
  for (int r = blockIdx.x; r < n; r += gridDim.x) s += tile[(long)r * HD + h];
  atomicAdd(&out[HD + h], s);
}

// =====================================================================================
extern "C" void kernel_launch(void* const* d_in, const int* in_sizes, int n_in,
                              void* d_out, int out_size, void* d_ws, size_t ws_size,
                              hipStream_t stream)
{
  (void)n_in; (void)out_size; (void)ws_size;
  const float* tileF = (const float*)d_in[0];
  const float* rrF   = (const float*)d_in[1];
  const int* t2t_s = (const int*)d_in[2];
  const int* t2t_d = (const int*)d_in[3];
  const int* r2t_s = (const int*)d_in[4];
  const int* r2t_d = (const int*)d_in[5];
  const int* t2r_s = (const int*)d_in[6];
  const int* t2r_d = (const int*)d_in[7];
  const int* sink  = (const int*)d_in[8];
  const int E_t2t = in_sizes[2], E_r2t = in_sizes[4], E_t2r = in_sizes[6];
  const int n_tile = in_sizes[0] / HD, n_rr = in_sizes[1] / HD;
  const int n_max = n_tile > n_rr ? n_tile : n_rr;
  int E_max = E_t2t; if (E_r2t > E_max) E_max = E_r2t; if (E_t2r > E_max) E_max = E_t2r;

  struct Mlp { const float *W1, *b1, *W2, *b2; };
  auto mlp = [&](int i) {
    return Mlp{ (const float*)d_in[9 + 4*i], (const float*)d_in[10 + 4*i],
                (const float*)d_in[11 + 4*i], (const float*)d_in[12 + 4*i] };
  };
  const float* poolw = (const float*)d_in[33];
  const float* poolb = (const float*)d_in[34];

  // ---- workspace carve-out ----
  char* ws = (char*)d_ws;
  size_t off = 0;
  auto alloc = [&](size_t bytes) { char* p = ws + off; off = (off + bytes + 255) & ~(size_t)255; return p; };
  float*     tileA  = (float*)alloc((size_t)n_tile * HD * 4);
  float*     tileB  = (float*)alloc((size_t)n_tile * HD * 4);
  float*     rrA    = (float*)alloc((size_t)n_rr * HD * 4);
  float*     rrB    = (float*)alloc((size_t)n_rr * HD * 4);
  _Float16*  srcH   = (_Float16*)alloc((size_t)n_max * HD * 2);
  _Float16*  dstH   = (_Float16*)alloc((size_t)n_max * HD * 2);
  float*     m_buf  = (float*)alloc((size_t)E_max * HD * 4);
  float*     negb   = (float*)alloc((size_t)E_max * 4);
  unsigned*  segmax = (unsigned*)alloc((size_t)n_max * 4);
  float*     denom  = (float*)alloc((size_t)n_max * 4);
  unsigned*  scores = (unsigned*)alloc((size_t)n_rr * 4);
  unsigned*  keep   = (unsigned*)alloc((size_t)n_rr * 4);
  unsigned*  counts = (unsigned*)alloc(256 * 4);
  unsigned*  sel    = (unsigned*)alloc(4 * 4);

  auto run_mp = [&](const float* sF, int n_src, const float* dF, int n_dst,
                    const int* si, const int* di, int E, Mlp P, float* outF) {
    cast_f16_k<<<(unsigned)(((long)n_src * HD + 255) / 256), 256, 0, stream>>>(sF, srcH, (long)n_src * HD);
    const _Float16* dh = srcH;
    if (dF != sF) {
      cast_f16_k<<<(unsigned)(((long)n_dst * HD + 255) / 256), 256, 0, stream>>>(dF, dstH, (long)n_dst * HD);
      dh = dstH;
    }
    zero_seg_k<<<(unsigned)(((long)n_dst * HD + 255) / 256), 256, 0, stream>>>(outF, denom, segmax, n_dst);
    edge_mlp_pass1<<<(unsigned)((E + 127) / 128), 256, 0, stream>>>(
        srcH, dh, si, di, P.W1, P.b1, P.W2, P.b2, m_buf, negb, segmax, E);
    edge_scatter<<<(unsigned)(((long)E * 32 + 255) / 256), 256, 0, stream>>>(
        m_buf, negb, segmax, di, outF, denom, E);
    normalize_k<<<(unsigned)(((long)n_dst * HD + 255) / 256), 256, 0, stream>>>(outF, denom, n_dst);
  };

  // ---- round 1..3 ----
  run_mp(tileF, n_tile, tileF, n_tile, t2t_s, t2t_d, E_t2t, mlp(0), tileA); // t2t
  run_mp(rrF,   n_rr,   tileA, n_tile, r2t_s, r2t_d, E_r2t, mlp(1), tileB); // r2t
  run_mp(tileB, n_tile, rrF,   n_rr,   t2r_s, t2r_d, E_t2r, mlp(2), rrA);   // t2r

  // ---- TopK pool on rrA -> rrB ----
  pool_score<<<(unsigned)((n_rr + 255) / 256), 256, 0, stream>>>(rrA, poolw, poolb, scores, n_rr);
  unsigned K = (unsigned)(n_rr / 2); if (K < 1u) K = 1u;
  sel_init<<<1, 256, 0, stream>>>(counts, sel, K);
  for (int lvl = 0; lvl < 4; ++lvl) {
    int shift = 24 - 8 * lvl;
    unsigned hi_mask = (lvl == 0) ? 0u : (0xFFFFFFFFu << (shift + 8));
    topk_hist<<<(unsigned)((n_rr + 255) / 256), 256, 0, stream>>>(scores, counts, sel, n_rr, shift, hi_mask);
    topk_select<<<1, 1, 0, stream>>>(counts, sel, shift);
  }
  topk_flags<<<(unsigned)((n_rr + 255) / 256), 256, 0, stream>>>(scores, sel, keep, n_rr);
  topk_apply<<<(unsigned)(((long)n_rr * HD + 255) / 256), 256, 0, stream>>>(rrA, keep, rrB, n_rr);

  // ---- round 4..6 ----
  run_mp(tileB, n_tile, tileB, n_tile, t2t_s, t2t_d, E_t2t, mlp(3), tileA); // pt2t
  run_mp(rrB,   n_rr,   tileA, n_tile, r2t_s, r2t_d, E_r2t, mlp(4), tileB); // pr2t
  run_mp(tileB, n_tile, rrB,   n_rr,   t2r_s, t2r_d, E_t2r, mlp(5), rrA);   // pt2r

  // ---- readout ----
  readout_init<<<1, 256, 0, stream>>>((float*)d_out, rrA, sink);
  readout_sum<<<256, HD, 0, stream>>>(tileB, (float*)d_out, n_tile);
}
